// AttnBlock_90787018703337
// MI455X (gfx1250) — compile-verified, matrix operand order checked by reading, not run on hardware
//
#include <hip/hip_runtime.h>

typedef unsigned int u32;
typedef unsigned short u16;
typedef __attribute__((ext_vector_type(16))) __bf16 v16bf;
typedef __attribute__((ext_vector_type(8)))  float  v8f;
typedef __attribute__((ext_vector_type(8)))  u32    v8u;
typedef __attribute__((ext_vector_type(4)))  u32    v4u;

// ---------- helpers ----------
__device__ inline u16 f2bf(float f) {
    u32 u = __builtin_bit_cast(u32, f);
    u32 r = u + 0x7fffu + ((u >> 16) & 1u);   // round-to-nearest-even
    return (u16)(r >> 16);
}
__device__ inline u32 pack_bf2(float lo, float hi) {
    return (u32)f2bf(lo) | ((u32)f2bf(hi) << 16);
}
// Fragment load: ISA 7.12.2 16-bit A/B layout. For lane (l16, half) the 8 bf16
// pairs live at K = k0+8*half+[0..7] and K = k0+16+8*half+[0..7] -> two 16-byte
// contiguous chunks => two b128 loads (works for global and LDS pointers).
__device__ inline v16bf load_frag(const u32* base, int row, int stride_e,
                                  int k0, int half) {
    const v4u* p = (const v4u*)(base + (((size_t)row * stride_e + k0 + 8 * half) >> 1));
    v4u lo = p[0];
    v4u hi = p[2];                 // +16 bf16 elements = +8 u32 = +2 v4u
    v8u u;
    u[0] = lo[0]; u[1] = lo[1]; u[2] = lo[2]; u[3] = lo[3];
    u[4] = hi[0]; u[5] = hi[1]; u[6] = hi[2]; u[7] = hi[3];
    return __builtin_bit_cast(v16bf, u);
}
__device__ inline v8f wmma_bf16(v16bf a, v16bf b, v8f c) {
    return __builtin_amdgcn_wmma_f32_16x16x32_bf16(false, a, false, b, (short)0, c,
                                                   false, false);
}
// CDNA5 async global->LDS copy (16B per lane), tracked by ASYNCcnt (ISA ch10).
__device__ inline void async_b128(u32 lds_off, const void* gaddr) {
    asm volatile("global_load_async_to_lds_b128 %0, %1, off"
                 :: "v"(lds_off), "v"(gaddr) : "memory");
}
__device__ inline void wait_async0() {
    asm volatile("s_wait_asynccnt 0x0" ::: "memory");
}

// ---------- 1) GroupNorm statistics: one block per (b, g) ----------
__global__ __launch_bounds__(256) void gn_stats(const float* __restrict__ x,
                                                float* __restrict__ mean,
                                                float* __restrict__ rstd) {
    int bg = blockIdx.x;            // 0..255
    int b = bg >> 5, g = bg & 31;
    const float* xb = x + (size_t)b * 4096 * 256 + g * 8;
    float s = 0.f, ss = 0.f;
    for (int i = threadIdx.x; i < 4096 * 4; i += 256) {   // float2 per step
        int pix = i >> 2, chp = (i & 3) << 1;
        float2 v = *(const float2*)(xb + (size_t)pix * 256 + chp);
        s += v.x + v.y; ss += v.x * v.x + v.y * v.y;
    }
    __shared__ float rs[256], rq[256];
    rs[threadIdx.x] = s; rq[threadIdx.x] = ss;
    __syncthreads();
    for (int off = 128; off > 0; off >>= 1) {
        if ((int)threadIdx.x < off) {
            rs[threadIdx.x] += rs[threadIdx.x + off];
            rq[threadIdx.x] += rq[threadIdx.x + off];
        }
        __syncthreads();
    }
    if (threadIdx.x == 0) {
        float m = rs[0] * (1.f / 32768.f);
        float var = rq[0] * (1.f / 32768.f) - m * m;
        mean[bg] = m;
        rstd[bg] = rsqrtf(var + 1e-6f);
    }
}

// ---------- 2) normalize + affine, emit bf16 hnorm [32768][256] ----------
__global__ __launch_bounds__(256) void gn_apply(const float* __restrict__ x,
                                                const float* __restrict__ gs,
                                                const float* __restrict__ gb,
                                                const float* __restrict__ mean,
                                                const float* __restrict__ rstd,
                                                u32* __restrict__ hn) {
    int i = blockIdx.x * 256 + threadIdx.x;          // bf16-pair index
    int m = i >> 7;                                   // row (0..32767)
    int chp = (i & 127) << 1;                         // even channel
    int bg = ((m >> 12) << 5) | (chp >> 3);
    float2 xv = *(const float2*)(x + ((size_t)m << 8) + chp);
    float mu = mean[bg], rc = rstd[bg];
    float a = (xv.x - mu) * rc * gs[chp] + gb[chp];
    float c = (xv.y - mu) * rc * gs[chp + 1] + gb[chp + 1];
    hn[(size_t)m * 128 + (chp >> 1)] = pack_bf2(a, c);
}

// ---------- 3) all 4 weights: f32 [K][N] -> bf16 transposed [N][K] ----------
__global__ __launch_bounds__(256) void wconv4(const float* __restrict__ w0,
                                              const float* __restrict__ w1,
                                              const float* __restrict__ w2,
                                              const float* __restrict__ w3,
                                              u16* __restrict__ t0,
                                              u16* __restrict__ t1,
                                              u16* __restrict__ t2,
                                              u16* __restrict__ t3) {
    int i = blockIdx.x * 256 + threadIdx.x;           // 4*65536
    int which = i >> 16, e = i & 65535;
    int k = e >> 8, n = e & 255;
    const float* w = (which == 0) ? w0 : (which == 1) ? w1 : (which == 2) ? w2 : w3;
    u16* t = (which == 0) ? t0 : (which == 1) ? t1 : (which == 2) ? t2 : t3;
    t[n * 256 + k] = f2bf(w[e]);
}

// ---------- 4) fused QKV GEMM (bf16 WMMA), V stored transposed ----------
__global__ __launch_bounds__(256) void qkv_gemm(const u32* __restrict__ hn,
                                                const u32* __restrict__ wqt,
                                                const u32* __restrict__ wkt,
                                                const u32* __restrict__ wvt,
                                                const float* __restrict__ bq,
                                                const float* __restrict__ bk,
                                                const float* __restrict__ bv,
                                                u16* __restrict__ qo,
                                                u16* __restrict__ ko,
                                                u16* __restrict__ vto) {
    int wave = (blockIdx.x * 256 + threadIdx.x) >> 5;
    int lane = threadIdx.x & 31;
    int mt = wave >> 2, nq = wave & 3;                 // 16-row tile, 64-col strip
    int half = lane >> 4, l16 = lane & 15;
    int row = mt * 16 + l16;
    v8f aq[4] = {}, ak[4] = {}, av[4] = {};
#pragma unroll
    for (int ks = 0; ks < 8; ++ks) {
        int k0 = ks * 32;
        v16bf a = load_frag(hn, row, 256, k0, half);
#pragma unroll
        for (int t = 0; t < 4; ++t) {
            int n = nq * 64 + t * 16 + l16;
            v16bf fq = load_frag(wqt, n, 256, k0, half);
            v16bf fk = load_frag(wkt, n, 256, k0, half);
            v16bf fv = load_frag(wvt, n, 256, k0, half);
            aq[t] = wmma_bf16(a, fq, aq[t]);
            ak[t] = wmma_bf16(a, fk, ak[t]);
            av[t] = wmma_bf16(a, fv, av[t]);
        }
    }
#pragma unroll
    for (int t = 0; t < 4; ++t) {
        int n = nq * 64 + t * 16 + l16;
        float cq = bq[n], ck = bk[n], cv = bv[n];
#pragma unroll
        for (int r = 0; r < 8; ++r) {
            int m = mt * 16 + r + 8 * half;
            // fold softmax scale c^-0.5 = 1/16 into Q
            qo[(size_t)m * 256 + n] = f2bf((aq[t][r] + cq) * 0.0625f);
            ko[(size_t)m * 256 + n] = f2bf(ak[t][r] + ck);
            int b = m >> 12;
            vto[((size_t)(b * 256 + n) << 12) + (m & 4095)] = f2bf(av[t][r] + cv);
        }
    }
}

// ---------- 5) flash attention with async-LDS K/V staging ----------
// Block = 4 waves, each owns one 16-query block; all 4 share batch b.
// K/V chunks (32 keys) are double-buffered in LDS, staged cooperatively with
// GLOBAL_LOAD_ASYNC_TO_LDS_B128 while WMMAs consume the previous chunk.
__global__ __launch_bounds__(128) void attn(const u32* __restrict__ qg,
                                            const u32* __restrict__ kg,
                                            const u32* __restrict__ vtg,
                                            u16* __restrict__ og) {
    extern __shared__ char smem[];
    u32* qs = (u32*)smem;                    // 4 waves * 2048 u32 (Q 16x256) 32KB
    u16* ps = (u16*)(smem + 32768);          // 4 waves * 512 bf16 (P 16x32)   4KB
    u32* kv = (u32*)(smem + 36864);          // 2 bufs * (4096 K + 4096 V)    64KB

    int wv = threadIdx.x >> 5, lane = threadIdx.x & 31;
    int tid = threadIdx.x;
    int gw = blockIdx.x * 4 + wv;      // 0..2047
    int b = gw >> 8, qb = gw & 255;
    int m0 = qb * 16;
    int half = lane >> 4, l16 = lane & 15;

    u32* qsw = qs + wv * 2048;
    u16* psw = ps + wv * 512;
    const u32* kbb = kg + ((size_t)b << 19);    // b*4096*128 words
    const u32* vtb = vtg + ((size_t)b << 19);   // b*256*2048 words

    // async-stage this wave's Q tile (16 rows x 128 words, contiguous)
    {
        const char* qsrc = (const char*)(qg + ((size_t)(b * 4096 + m0) << 7));
        u32 qdst = (u32)(uintptr_t)qsw;
#pragma unroll
        for (int j = 0; j < 16; ++j) {
            int t = j * 32 + lane;           // 512 x 16B
            async_b128(qdst + t * 16, qsrc + t * 16);
        }
    }
    // stage K/V chunk kb into LDS buffer (K: [32 keys][256ch], V: [256ch][32seq])
    u32 kv0 = (u32)(uintptr_t)kv;
    auto stage = [&](int buf, int kb) {
        u32 kdst = kv0 + buf * 32768;
        u32 vdst = kdst + 16384;
        const char* ksrc = (const char*)(kbb + (size_t)kb * 4096);
#pragma unroll
        for (int j = 0; j < 8; ++j) {
            int t = j * 128 + tid;           // 1024 x 16B each for K and V
            async_b128(kdst + t * 16, ksrc + t * 16);
            int ch = t >> 2, wi = t & 3;
            async_b128(vdst + t * 16,
                       (const char*)(vtb + (size_t)ch * 2048 + kb * 16 + wi * 4));
        }
    };
    stage(0, 0);

    float mrow[8], lrow[8];
#pragma unroll
    for (int r = 0; r < 8; ++r) { mrow[r] = -3.0e38f; lrow[r] = 0.f; }
    v8f acc[16] = {};

    for (int kb = 0; kb < 128; ++kb) {
        wait_async0();                 // our share of current buffer landed
        __syncthreads();               // whole block's staging visible
        const u32* kbuf = (const u32*)(uintptr_t)(size_t)0;
        u32 cur = kv0 + (kb & 1) * 32768;
        kbuf = kv + (kb & 1) * 8192;
        const u32* vbuf = kbuf + 4096;
        (void)cur;
        if (kb < 127) stage((kb + 1) & 1, kb + 1);   // overlap next staging

        v8f s0 = {}, s1 = {};
        // software-pipelined score loop over K=256 (all operands from LDS)
        v16bf a  = load_frag(qsw, l16, 256, 0, half);
        v16bf f0 = load_frag(kbuf, l16, 256, 0, half);
        v16bf f1 = load_frag(kbuf, 16 + l16, 256, 0, half);
#pragma unroll
        for (int ks = 0; ks < 8; ++ks) {
            v16bf an = a, b0 = f0, b1 = f1;
            if (ks < 7) {
                int k0 = (ks + 1) * 32;
                a  = load_frag(qsw, l16, 256, k0, half);
                f0 = load_frag(kbuf, l16, 256, k0, half);
                f1 = load_frag(kbuf, 16 + l16, 256, k0, half);
            }
            s0 = wmma_bf16(an, b0, s0);
            s1 = wmma_bf16(an, b1, s1);
        }
        float scale[8];
#pragma unroll
        for (int r = 0; r < 8; ++r) {
            float v = fmaxf(s0[r], s1[r]);
            v = fmaxf(v, __shfl_xor(v, 1, 32));
            v = fmaxf(v, __shfl_xor(v, 2, 32));
            v = fmaxf(v, __shfl_xor(v, 4, 32));
            v = fmaxf(v, __shfl_xor(v, 8, 32));     // row max within 16-lane half
            float mn = fmaxf(mrow[r], v);
            scale[r] = __expf(mrow[r] - mn);
            mrow[r] = mn;
            float p0 = __expf(s0[r] - mn);
            float p1 = __expf(s1[r] - mn);
            int prow = r + 8 * half;
            psw[prow * 32 + l16] = f2bf(p0);
            psw[prow * 32 + 16 + l16] = f2bf(p1);
            float rsum = p0 + p1;
            rsum += __shfl_xor(rsum, 1, 32);
            rsum += __shfl_xor(rsum, 2, 32);
            rsum += __shfl_xor(rsum, 4, 32);
            rsum += __shfl_xor(rsum, 8, 32);
            lrow[r] = lrow[r] * scale[r] + rsum;
        }
#pragma unroll
        for (int t = 0; t < 16; ++t)
#pragma unroll
            for (int r = 0; r < 8; ++r) acc[t][r] = acc[t][r] * scale[r];
        // P tile read back from LDS in A-fragment layout (two b128 reads)
        v16bf ap = load_frag((const u32*)psw, l16, 32, 0, half);
        // software-pipelined P*V accumulation over 256 output channels
        v16bf fv = load_frag(vbuf, l16, 32, 0, half);
#pragma unroll
        for (int t = 0; t < 16; ++t) {
            v16bf bt = fv;
            if (t < 15)
                fv = load_frag(vbuf, (t + 1) * 16 + l16, 32, 0, half);
            acc[t] = wmma_bf16(ap, bt, acc[t]);
        }
    }
    float inv[8];
#pragma unroll
    for (int r = 0; r < 8; ++r) inv[r] = 1.f / lrow[r];
    u16* orow = og + ((size_t)(b * 4096 + m0) << 8);
#pragma unroll
    for (int t = 0; t < 16; ++t)
#pragma unroll
        for (int r = 0; r < 8; ++r)
            orow[(size_t)(r + 8 * half) * 256 + t * 16 + l16] =
                f2bf(acc[t][r] * inv[r]);
}

// ---------- 6) output projection + bias + residual ----------
__global__ __launch_bounds__(256) void proj_out(const u32* __restrict__ ao,
                                                const u32* __restrict__ wpt,
                                                const float* __restrict__ bp,
                                                const float* __restrict__ xin,
                                                float* __restrict__ out) {
    int wave = (blockIdx.x * 256 + threadIdx.x) >> 5;
    int lane = threadIdx.x & 31;
    int mt = wave >> 2, nq = wave & 3;
    int half = lane >> 4, l16 = lane & 15;
    int row = mt * 16 + l16;
    v8f acc[4] = {};
#pragma unroll
    for (int ks = 0; ks < 8; ++ks) {
        int k0 = ks * 32;
        v16bf a = load_frag(ao, row, 256, k0, half);
#pragma unroll
        for (int t = 0; t < 4; ++t) {
            int n = nq * 64 + t * 16 + l16;
            acc[t] = wmma_bf16(a, load_frag(wpt, n, 256, k0, half), acc[t]);
        }
    }
#pragma unroll
    for (int t = 0; t < 4; ++t) {
        int n = nq * 64 + t * 16 + l16;
        float c = bp[n];
#pragma unroll
        for (int r = 0; r < 8; ++r) {
            size_t idx = ((size_t)(mt * 16 + r + 8 * half) << 8) + n;
            out[idx] = xin[idx] + acc[t][r] + c;
        }
    }
}

extern "C" void kernel_launch(void* const* d_in, const int* in_sizes, int n_in,
                              void* d_out, int out_size, void* d_ws, size_t ws_size,
                              hipStream_t stream) {
    (void)in_sizes; (void)n_in; (void)out_size; (void)ws_size;
    const float* x  = (const float*)d_in[0];
    const float* gs = (const float*)d_in[1];
    const float* gb = (const float*)d_in[2];
    const float* wq = (const float*)d_in[3];
    const float* bq = (const float*)d_in[4];
    const float* wk = (const float*)d_in[5];
    const float* bk = (const float*)d_in[6];
    const float* wvv= (const float*)d_in[7];
    const float* bv = (const float*)d_in[8];
    const float* wp = (const float*)d_in[9];
    const float* bp = (const float*)d_in[10];
    float* out = (float*)d_out;

    char* ws = (char*)d_ws;
    const size_t SZ_HN = (size_t)32768 * 256 * 2;   // bf16 matrices
    const size_t SZ_W  = (size_t)256 * 256 * 2;
    float* mean = (float*)(ws);                     //  1 KB
    float* rstd = (float*)(ws + 1024);              //  1 KB
    char*  hn   = ws + 4096;
    char*  wqt  = hn  + SZ_HN;
    char*  wkt  = wqt + SZ_W;
    char*  wvt  = wkt + SZ_W;
    char*  wpt  = wvt + SZ_W;
    char*  q    = wpt + SZ_W;
    char*  k    = q   + SZ_HN;
    char*  vt   = k   + SZ_HN;
    char*  ao   = vt  + SZ_HN;                      // total ~84.4 MB

    gn_stats<<<256, 256, 0, stream>>>(x, mean, rstd);
    gn_apply<<<16384, 256, 0, stream>>>(x, gs, gb, mean, rstd, (u32*)hn);
    wconv4<<<1024, 256, 0, stream>>>(wq, wk, wvv, wp,
                                     (u16*)wqt, (u16*)wkt, (u16*)wvt, (u16*)wpt);
    qkv_gemm<<<1024, 256, 0, stream>>>((const u32*)hn, (const u32*)wqt,
                                       (const u32*)wkt, (const u32*)wvt,
                                       bq, bk, bv, (u16*)q, (u16*)k, (u16*)vt);
    attn<<<512, 128, 102400, stream>>>((const u32*)q, (const u32*)k,
                                       (const u32*)vt, (u16*)ao);
    proj_out<<<1024, 256, 0, stream>>>((const u32*)ao, (const u32*)wpt, bp, x, out);
}
